// GraphNN_9955734192168
// MI455X (gfx1250) — compile-verified, matrix operand order checked by reading, not run on hardware
//
#include <hip/hip_runtime.h>
#include <hip/hip_bf16.h>

// ---------------------------------------------------------------------------
// GNN (3x TransformerConv, H=64) for MI455X / gfx1250, wave32 + WMMA bf16.
// All dense GEMMs run on v_wmma_f32_16x16x32_bf16 (bf16 in, f32 accumulate).
// B-fragments (weights) are staged in LDS per 8-wave block and fed from
// ds_load_b128; A-fragments built in-register from f32 rows (RNE -> bf16).
// ---------------------------------------------------------------------------

#define N_NODES 50000
#define N_EDGES 800000
#define HD      64

typedef __attribute__((ext_vector_type(16))) __bf16 v16bf;
typedef __attribute__((ext_vector_type(8)))  float  v8f;

union BF16Frag {
    v16bf          v;
    unsigned short u[16];
    uint4          q[2];
};

__device__ __forceinline__ unsigned short f2bf(float f) {
    unsigned int u = __float_as_uint(f);
    u += 0x7FFFu + ((u >> 16) & 1u);        // round-to-nearest-even
    return (unsigned short)(u >> 16);
}

// A-fragment (16x32 bf16, wave32) per ISA 7.12.2: lane L holds row m=L%16,
// half=L/16; shorts j=0..7 -> K = kb + half*8 + j, j=8..15 -> K = kb+16+half*8+(j-8).
__device__ __forceinline__ void load_a_frag(BF16Frag& fr, const float* row, int kb, int half) {
    const float* p0 = row + kb + half * 8;
    const float* p1 = p0 + 16;
    float4 a = *(const float4*)(p0);
    float4 b = *(const float4*)(p0 + 4);
    float4 c = *(const float4*)(p1);
    float4 d = *(const float4*)(p1 + 4);
    fr.u[0]  = f2bf(a.x); fr.u[1]  = f2bf(a.y); fr.u[2]  = f2bf(a.z); fr.u[3]  = f2bf(a.w);
    fr.u[4]  = f2bf(b.x); fr.u[5]  = f2bf(b.y); fr.u[6]  = f2bf(b.z); fr.u[7]  = f2bf(b.w);
    fr.u[8]  = f2bf(c.x); fr.u[9]  = f2bf(c.y); fr.u[10] = f2bf(c.z); fr.u[11] = f2bf(c.w);
    fr.u[12] = f2bf(d.x); fr.u[13] = f2bf(d.y); fr.u[14] = f2bf(d.z); fr.u[15] = f2bf(d.w);
}

// B-fragment from LDS-staged, frag-major swizzled weights: one lane's 16
// bf16 elements are 32 contiguous bytes -> two ds_load_b128 per fragment.
__device__ __forceinline__ void load_b_frag(BF16Frag& fr, const unsigned short* w, int frag, int lane) {
    const uint4* p = (const uint4*)(w + ((size_t)(frag * 32 + lane) << 4));
    fr.q[0] = p[0];
    fr.q[1] = p[1];
}

__device__ __forceinline__ v8f wmma_bf16(const BF16Frag& a, const BF16Frag& b, v8f c) {
    return __builtin_amdgcn_wmma_f32_16x16x32_bf16(false, a.v, false, b.v, (short)0, c, false, false);
}

// cooperative global->LDS staging of swizzled weights (nshorts % (8*blockDim) == 0 not required)
__device__ __forceinline__ void stage_weights(const unsigned short* __restrict__ g,
                                              unsigned short* s, int nshorts) {
    const uint4* gs = (const uint4*)g;
    uint4*       ss = (uint4*)s;
    int n = nshorts >> 3;                    // uint4 = 8 shorts
    for (int i = threadIdx.x; i < n; i += blockDim.x) ss[i] = gs[i];
}

__device__ __forceinline__ void atomic_max_float(float* addr, float val) {
    // sign-split: IEEE order == signed-int order for >=0, reversed-unsigned for <0
    if (val >= 0.0f) atomicMax((int*)addr, __float_as_int(val));
    else             atomicMin((unsigned int*)addr, __float_as_uint(val));
}

// --------------------------------------------------------------------------
// Weight swizzle: f32 [K x 64] row-major -> bf16 B-fragment-major layout.
// --------------------------------------------------------------------------
__global__ void swizzle_weight_kernel(const float* __restrict__ W,
                                      unsigned short* __restrict__ out, int K) {
    int tid = blockIdx.x * 256 + threadIdx.x;
    if (tid >= K * HD) return;
    int j     = tid & 15;
    int lane  = (tid >> 4) & 31;
    int f     = tid >> 9;
    int kstep = f >> 2, t = f & 3;
    int half  = lane >> 4, ln = lane & 15;
    int r     = j >> 1, lohi = j & 1;
    int k;
    if (r < 4) k = kstep * 32 + half * 8 + 2 * r + lohi;
    else       k = kstep * 32 + 16 + half * 8 + 2 * (r - 4) + lohi;
    int n = t * 16 + ln;
    out[tid] = f2bf(W[(size_t)k * HD + n]);
}

// --------------------------------------------------------------------------
// Input projections (tiny K: scalar FMA, memory bound anyway)
// --------------------------------------------------------------------------
__global__ void node_proj_kernel(const float* __restrict__ x, const float* __restrict__ Wnp,
                                 const float* __restrict__ bnp, float* __restrict__ h) {
    size_t t = (size_t)blockIdx.x * 256 + threadIdx.x;
    if (t >= (size_t)N_NODES * HD) return;
    size_t node = t >> 6; int c = (int)(t & 63);
    const float* xr = x + node * 16;
    float acc = bnp[c];
#pragma unroll
    for (int i = 0; i < 16; ++i) acc += xr[i] * Wnp[i * HD + c];
    h[t] = acc;
}

__global__ void edge_proj_kernel(const float* __restrict__ ea, const float* __restrict__ Wep,
                                 const float* __restrict__ bep, float* __restrict__ e) {
    size_t t = (size_t)blockIdx.x * 256 + threadIdx.x;
    if (t >= (size_t)N_EDGES * HD) return;
    size_t edge = t >> 6; int c = (int)(t & 63);
    const float* er = ea + edge * 8;
    float acc = bep[c];
#pragma unroll
    for (int i = 0; i < 8; ++i) acc += er[i] * Wep[i * HD + c];
    e[t] = acc;
}

// --------------------------------------------------------------------------
// q/k/v/skip: per wave, 4 GEMMs [16 nodes x 64] x [64 x 64], 32 WMMAs.
// 8 waves / block share 32KB of LDS-staged weights.
// --------------------------------------------------------------------------
__device__ __forceinline__ void wave_gemm_k64_store(const BF16Frag& a0, const BF16Frag& a1,
                                                    const unsigned short* Wlds,
                                                    const float* __restrict__ bias,
                                                    float* __restrict__ out,
                                                    size_t row_base, int lane) {
    int half = lane >> 4, ln = lane & 15;
#pragma unroll
    for (int t = 0; t < 4; ++t) {
        BF16Frag b0, b1;
        load_b_frag(b0, Wlds, 0 * 4 + t, lane);
        load_b_frag(b1, Wlds, 1 * 4 + t, lane);
        v8f c = {};
        c = wmma_bf16(a0, b0, c);
        c = wmma_bf16(a1, b1, c);
        int n = t * 16 + ln;
        float bb = bias[n];
#pragma unroll
        for (int r = 0; r < 8; ++r) {
            size_t row = row_base + (size_t)(r + 8 * half);
            out[row * HD + n] = c[r] + bb;
        }
    }
}

__global__ void __launch_bounds__(256)
node_qkvs_kernel(const float* __restrict__ h, const unsigned short* __restrict__ Wsw,
                 const float* __restrict__ bq, const float* __restrict__ bk,
                 const float* __restrict__ bv, const float* __restrict__ bs,
                 float* __restrict__ q, float* __restrict__ k,
                 float* __restrict__ v, float* __restrict__ hs) {
    __shared__ unsigned short sW[4 * 4096];          // Wq,Wk,Wv,Ws swizzled (32KB)
    stage_weights(Wsw, sW, 4 * 4096);
    __syncthreads();

    int lane = threadIdx.x & 31;
    int wave = threadIdx.x >> 5;
    size_t nb = (size_t)blockIdx.x * 128 + (size_t)wave * 16;
    if (nb >= (size_t)N_NODES) return;               // wave-uniform (N%16==0)

    int half = lane >> 4, ln = lane & 15;
    const float* row = h + (nb + ln) * HD;
    BF16Frag a0, a1;
    load_a_frag(a0, row, 0, half);
    load_a_frag(a1, row, 32, half);
    wave_gemm_k64_store(a0, a1, sW + 0 * 4096, bq, q,  nb, lane);
    wave_gemm_k64_store(a0, a1, sW + 1 * 4096, bk, k,  nb, lane);
    wave_gemm_k64_store(a0, a1, sW + 2 * 4096, bv, v,  nb, lane);
    wave_gemm_k64_store(a0, a1, sW + 3 * 4096, bs, hs, nb, lane);
}

// --------------------------------------------------------------------------
// ee = e @ We (WMMA), fused: key_e = ee + k[src], val_e = ee + v[src].
// 8 waves / block, 128 edges / block, We staged in 8KB LDS.
// --------------------------------------------------------------------------
__global__ void __launch_bounds__(256)
edge_attn_prep_kernel(const float* __restrict__ e, const unsigned short* __restrict__ We_sw,
                      const float* __restrict__ kbuf, const float* __restrict__ vbuf,
                      const int* __restrict__ src,
                      float* __restrict__ key_e, float* __restrict__ val_e) {
    __shared__ unsigned short sW[4096];              // We swizzled (8KB)
    stage_weights(We_sw, sW, 4096);
    __syncthreads();

    int lane = threadIdx.x & 31;
    int wave = threadIdx.x >> 5;
    size_t eb = (size_t)blockIdx.x * 128 + (size_t)wave * 16;

    int half = lane >> 4, ln = lane & 15;
    const float* row = e + (eb + ln) * HD;
    BF16Frag a0, a1;
    load_a_frag(a0, row, 0, half);
    load_a_frag(a1, row, 32, half);
#pragma unroll
    for (int t = 0; t < 4; ++t) {
        BF16Frag b0, b1;
        load_b_frag(b0, sW, 0 * 4 + t, lane);
        load_b_frag(b1, sW, 1 * 4 + t, lane);
        v8f c = {};
        c = wmma_bf16(a0, b0, c);
        c = wmma_bf16(a1, b1, c);
        int n = t * 16 + ln;
#pragma unroll
        for (int r = 0; r < 8; ++r) {
            size_t edge = eb + (size_t)(r + 8 * half);
            int si = src[edge];
            float ee = c[r];
            key_e[edge * HD + n] = ee + kbuf[(size_t)si * HD + n];
            val_e[edge * HD + n] = ee + vbuf[(size_t)si * HD + n];
        }
    }
}

// --------------------------------------------------------------------------
// scatter softmax pieces
// --------------------------------------------------------------------------
__global__ void attn_init_kernel(float* __restrict__ agg, float* __restrict__ m,
                                 float* __restrict__ s) {
    size_t t = (size_t)blockIdx.x * 256 + threadIdx.x;
    if (t < (size_t)N_NODES * HD) agg[t] = 0.0f;
    if (t < (size_t)N_NODES) { m[t] = __int_as_float(0xFF800000); s[t] = 0.0f; }
}

__global__ void alpha_max_kernel(const float* __restrict__ q, const float* __restrict__ key_e,
                                 const int* __restrict__ dst, float* __restrict__ alpha,
                                 float* __restrict__ m) {
    int e = blockIdx.x * 256 + threadIdx.x;
    if (e >= N_EDGES) return;
    if (e + 4096 < N_EDGES)
        __builtin_prefetch(key_e + (size_t)(e + 4096) * HD, 0, 0);  // global_prefetch_b8
    int d = dst[e];
    const float4* qr = (const float4*)(q + (size_t)d * HD);
    const float4* kr = (const float4*)(key_e + (size_t)e * HD);
    float acc = 0.0f;
#pragma unroll
    for (int i = 0; i < 16; ++i) {
        float4 a = qr[i], b = kr[i];
        acc += a.x * b.x + a.y * b.y + a.z * b.z + a.w * b.w;
    }
    acc *= 0.125f;              // 1/sqrt(64)
    alpha[e] = acc;
    atomic_max_float(m + d, acc);
}

__global__ void exp_sum_kernel(const int* __restrict__ dst, const float* __restrict__ m,
                               float* __restrict__ alpha, float* __restrict__ s) {
    int e = blockIdx.x * 256 + threadIdx.x;
    if (e >= N_EDGES) return;
    int d = dst[e];
    float ex = __expf(alpha[e] - m[d]);
    alpha[e] = ex;              // reuse buffer as "ex"
    atomicAdd(s + d, ex);
}

__global__ void aggregate_kernel(const int* __restrict__ dst, const float* __restrict__ ex,
                                 const float* __restrict__ s, const float* __restrict__ val_e,
                                 float* __restrict__ agg) {
    size_t t = (size_t)blockIdx.x * 256 + threadIdx.x;   // E*64 threads
    size_t e = t >> 6; int n = (int)(t & 63);
    int d = dst[e];
    float a = ex[e] / (s[d] + 1e-16f);
    atomicAdd(agg + (size_t)d * HD + n, a * val_e[e * HD + n]);
}

__global__ void node_finalize_kernel(const float* __restrict__ agg, const float* __restrict__ hs,
                                     float* __restrict__ h, int relu) {
    size_t t = (size_t)blockIdx.x * 256 + threadIdx.x;
    if (t >= (size_t)N_NODES * HD) return;
    float x = agg[t] + hs[t];
    if (relu) x = x > 0.0f ? x : 0.01f * x;
    h[t] = x;
}

// --------------------------------------------------------------------------
// edge update: e_out = [h[src], h[dst], e] @ Wu + bu  ([E,192]x[192,64]).
// 8 waves / block, Wu staged in 24KB LDS; 24 WMMAs per wave of 16 edges.
// --------------------------------------------------------------------------
__global__ void __launch_bounds__(256)
edge_update_kernel(const float* __restrict__ h, const float* __restrict__ e,
                   const unsigned short* __restrict__ Wu_sw, const float* __restrict__ bu,
                   const int* __restrict__ src, const int* __restrict__ dst,
                   float* __restrict__ e_out, int relu) {
    __shared__ unsigned short sW[24 * 512];          // Wu swizzled (24KB)
    stage_weights(Wu_sw, sW, 24 * 512);
    __syncthreads();

    int lane = threadIdx.x & 31;
    int wave = threadIdx.x >> 5;
    size_t eb = (size_t)blockIdx.x * 128 + (size_t)wave * 16;

    int half = lane >> 4, ln = lane & 15;
    size_t me = eb + ln;
    int si = src[me], di = dst[me];
    BF16Frag a[6];
    load_a_frag(a[0], h + (size_t)si * HD, 0,  half);
    load_a_frag(a[1], h + (size_t)si * HD, 32, half);
    load_a_frag(a[2], h + (size_t)di * HD, 0,  half);
    load_a_frag(a[3], h + (size_t)di * HD, 32, half);
    load_a_frag(a[4], e + me * HD, 0,  half);
    load_a_frag(a[5], e + me * HD, 32, half);
#pragma unroll
    for (int t = 0; t < 4; ++t) {
        v8f c = {};
#pragma unroll
        for (int ks = 0; ks < 6; ++ks) {
            BF16Frag b;
            load_b_frag(b, sW, ks * 4 + t, lane);
            c = wmma_bf16(a[ks], b, c);
        }
        int n = t * 16 + ln;
        float bb = bu[n];
#pragma unroll
        for (int r = 0; r < 8; ++r) {
            size_t edge = eb + (size_t)(r + 8 * half);
            float x = c[r] + bb;
            if (relu) x = x > 0.0f ? x : 0.01f * x;
            e_out[edge * HD + n] = x;
        }
    }
}

__global__ void final_kernel(const float* __restrict__ e, const float* __restrict__ Wf,
                             const float* __restrict__ bfp, float* __restrict__ out) {
    int ed = blockIdx.x * 256 + threadIdx.x;
    if (ed >= N_EDGES) return;
    const float4* er = (const float4*)(e + (size_t)ed * HD);
    const float4* wr = (const float4*)Wf;
    float acc = bfp[0];
#pragma unroll
    for (int i = 0; i < 16; ++i) {
        float4 a = er[i], b = wr[i];
        acc += a.x * b.x + a.y * b.y + a.z * b.z + a.w * b.w;
    }
    out[ed] = 1.0f / (1.0f + __expf(-acc));
}

// --------------------------------------------------------------------------
// launch
// --------------------------------------------------------------------------
extern "C" void kernel_launch(void* const* d_in, const int* in_sizes, int n_in,
                              void* d_out, int out_size, void* d_ws, size_t ws_size,
                              hipStream_t stream) {
    (void)in_sizes; (void)n_in; (void)out_size; (void)ws_size;

    const float* x   = (const float*)d_in[0];
    const float* ea  = (const float*)d_in[1];
    const int*   src = (const int*)d_in[2];
    const int*   dst = src + N_EDGES;
    const float* Wnp = (const float*)d_in[3];
    const float* bnp = (const float*)d_in[4];
    const float* Wep = (const float*)d_in[5];
    const float* bep = (const float*)d_in[6];
    const float* Wq  = (const float*)d_in[7];
    const float* bq  = (const float*)d_in[8];
    const float* Wk  = (const float*)d_in[9];
    const float* bk  = (const float*)d_in[10];
    const float* Wv  = (const float*)d_in[11];
    const float* bv  = (const float*)d_in[12];
    const float* We  = (const float*)d_in[13];
    const float* Ws  = (const float*)d_in[14];
    const float* bs  = (const float*)d_in[15];
    const float* Wu  = (const float*)d_in[16];
    const float* bu  = (const float*)d_in[17];
    const float* Wf  = (const float*)d_in[18];
    const float* bfp = (const float*)d_in[19];
    float* out = (float*)d_out;

    // workspace carve (256B aligned)
    char* w = (char*)d_ws;
    auto carve = [&](size_t bytes) -> void* {
        void* p = (void*)w;
        w += (bytes + 255) & ~(size_t)255;
        return p;
    };
    const size_t NH = (size_t)N_NODES * HD * sizeof(float);
    const size_t EH = (size_t)N_EDGES * HD * sizeof(float);
    float* h    = (float*)carve(NH);
    float* qb   = (float*)carve(NH);
    float* kb   = (float*)carve(NH);
    float* vb   = (float*)carve(NH);
    float* hs   = (float*)carve(NH);
    float* agg  = (float*)carve(NH);
    float* eA   = (float*)carve(EH);
    float* eB   = (float*)carve(EH);
    float* valE = (float*)carve(EH);
    float* alph = (float*)carve((size_t)N_EDGES * sizeof(float));
    float* mbuf = (float*)carve((size_t)N_NODES * sizeof(float));
    float* sbuf = (float*)carve((size_t)N_NODES * sizeof(float));
    const int PER_LAYER_SW = 5 * 4096 + 12288;               // shorts per layer
    unsigned short* wsw = (unsigned short*)carve((size_t)3 * PER_LAYER_SW * sizeof(unsigned short));

    // 1) swizzle all weights to bf16 B-fragment layout
    for (int l = 0; l < 3; ++l) {
        unsigned short* base = wsw + (size_t)l * PER_LAYER_SW;
        swizzle_weight_kernel<<<16, 256, 0, stream>>>(Wq + (size_t)l * 4096,  base +     0, 64);
        swizzle_weight_kernel<<<16, 256, 0, stream>>>(Wk + (size_t)l * 4096,  base +  4096, 64);
        swizzle_weight_kernel<<<16, 256, 0, stream>>>(Wv + (size_t)l * 4096,  base +  8192, 64);
        swizzle_weight_kernel<<<16, 256, 0, stream>>>(Ws + (size_t)l * 4096,  base + 12288, 64);
        swizzle_weight_kernel<<<16, 256, 0, stream>>>(We + (size_t)l * 4096,  base + 16384, 64);
        swizzle_weight_kernel<<<48, 256, 0, stream>>>(Wu + (size_t)l * 12288, base + 20480, 192);
    }

    // 2) input projections
    node_proj_kernel<<<(N_NODES * HD) / 256, 256, 0, stream>>>(x, Wnp, bnp, h);
    edge_proj_kernel<<<(N_EDGES / 256) * HD, 256, 0, stream>>>(ea, Wep, bep, eA);

    // 3) three TransformerConv + edge-update layers (ping-pong eA/eB;
    //    key_e borrows the "other" e buffer, then edge_update overwrites it)
    const int NODE_BLOCKS = (N_NODES + 127) / 128;           // 8 waves x 16 rows
    const int EDGE_BLOCKS = N_EDGES / 128;                   // exact (800000/128)
    for (int l = 0; l < 3; ++l) {
        float* ecur = (l & 1) ? eB : eA;
        float* eoth = (l & 1) ? eA : eB;
        const unsigned short* swl = wsw + (size_t)l * PER_LAYER_SW;
        int relu = (l < 2) ? 1 : 0;

        node_qkvs_kernel<<<NODE_BLOCKS, 256, 0, stream>>>(
            h, swl, bq + l * HD, bk + l * HD, bv + l * HD, bs + l * HD, qb, kb, vb, hs);

        edge_attn_prep_kernel<<<EDGE_BLOCKS, 256, 0, stream>>>(
            ecur, swl + 16384, kb, vb, src, /*key_e=*/eoth, valE);

        attn_init_kernel<<<(N_NODES * HD) / 256, 256, 0, stream>>>(agg, mbuf, sbuf);
        alpha_max_kernel<<<N_EDGES / 256, 256, 0, stream>>>(qb, /*key_e=*/eoth, dst, alph, mbuf);
        exp_sum_kernel<<<N_EDGES / 256, 256, 0, stream>>>(dst, mbuf, alph, sbuf);
        aggregate_kernel<<<(N_EDGES / 256) * HD, 256, 0, stream>>>(dst, alph, sbuf, valE, agg);
        node_finalize_kernel<<<(N_NODES * HD) / 256, 256, 0, stream>>>(agg, hs, h, relu);

        edge_update_kernel<<<EDGE_BLOCKS, 256, 0, stream>>>(
            h, ecur, swl + 20480, bu + l * HD, src, dst, /*e_out=*/eoth, relu);
    }

    // after l=0: e in eB; l=1: eA; l=2: eB
    final_kernel<<<N_EDGES / 256, 256, 0, stream>>>(eB, Wf, bfp, out);
}
